// HED_SNN_STDB_40991167873456
// MI455X (gfx1250) — matrix operand order repaired and checked by reading, not compiled, not run on hardware
//
#include <hip/hip_runtime.h>
#include <hip/hip_bf16.h>

typedef __attribute__((ext_vector_type(16))) _Float16 v16h;
typedef __attribute__((ext_vector_type(8)))  _Float16 v8h;
typedef __attribute__((ext_vector_type(8)))  float    v8f;

union V16H { v16h v; v8h q[2]; _Float16 h[16]; };
union V8F  { v8f  v; float f[8]; };

#define LEAK 1.0f
#define THR  1.0f
#define TSTEPS 5

// ---------------------------------------------------------------------------
// elementwise kernels
// ---------------------------------------------------------------------------
__global__ void zero_f32_kernel(float* __restrict__ p, int n) {
    int i = blockIdx.x * blockDim.x + threadIdx.x;
    if (i < n) p[i] = 0.0f;
}

// PoissonGenerator -> NHWC f16 with channels padded 3 -> 32 (zeros)
// img, rnd are NCHW [3][16384]
__global__ void poisson_kernel(const float* __restrict__ img,
                               const float* __restrict__ rnd,
                               _Float16* __restrict__ out /* [16384][32] */) {
    int i = blockIdx.x * blockDim.x + threadIdx.x;   // over 16384*32
    if (i >= 16384 * 32) return;
    int c = i & 31;
    int p = i >> 5;
    float o = 0.0f;
    if (c < 3) {
        float im = img[c * 16384 + p];
        float s  = (im > 0.0f) ? 1.0f : ((im < 0.0f) ? -1.0f : 0.0f);
        o = (rnd[c * 16384 + p] <= fabsf(im)) ? s : 0.0f;
    }
    out[i] = (_Float16)o;
}

// spike + membrane pre-decay: out = (mem/THR - 1 > 0); mem = LEAK*mem - THR*out
__global__ void spike_update_kernel(float* __restrict__ mem,
                                    _Float16* __restrict__ out, int n) {
    int i = blockIdx.x * blockDim.x + threadIdx.x;
    if (i >= n) return;
    float m = mem[i];
    float o = ((m / THR - 1.0f) > 0.0f) ? 1.0f : 0.0f;
    out[i]  = (_Float16)o;
    mem[i]  = LEAK * m - THR * o;
}

// 2x2 max pool, NHWC f16: in [H][W][C] -> out [H/2][W/2][C]
__global__ void maxpool2_kernel(const _Float16* __restrict__ in,
                                _Float16* __restrict__ out,
                                int C, int H, int W) {
    int Ho = H >> 1, Wo = W >> 1;
    int n = Ho * Wo * C;
    int i = blockIdx.x * blockDim.x + threadIdx.x;
    if (i >= n) return;
    int c   = i % C;
    int pix = i / C;
    int y = pix / Wo, x = pix - y * Wo;
    const _Float16* b = in + ((size_t)(2 * y) * W + 2 * x) * C + c;
    float a0 = (float)b[0],                 a1 = (float)b[C];
    float a2 = (float)b[(size_t)W * C],     a3 = (float)b[(size_t)W * C + C];
    out[i] = (_Float16)fmaxf(fmaxf(a0, a1), fmaxf(a2, a3));
}

// 1x1 conv to 1 channel, accumulated: acc[p] += sum_c sw[c] * spk[p][c] (NHWC)
__global__ void side_acc_kernel(const _Float16* __restrict__ spk,
                                const float* __restrict__ sw,
                                float* __restrict__ acc, int C, int HW) {
    int p = blockIdx.x * blockDim.x + threadIdx.x;
    if (p >= HW) return;
    const _Float16* row = spk + (size_t)p * C;
    float s = 0.0f;
    for (int c = 0; c < C; ++c)
        s += sw[c] * (float)row[c];
    acc[p] += s;
}

// fw (O,ICs,3,3) f32 -> w2 f16 layout [9][O][ICd], zero-padded ICs -> ICd
__global__ void wtrans_kernel(const float* __restrict__ fw,
                              _Float16* __restrict__ w2,
                              int O, int ICs, int ICd) {
    int i = blockIdx.x * blockDim.x + threadIdx.x;   // over 9*O*ICd
    if (i >= 9 * O * ICd) return;
    int ic = i % ICd;
    int t2 = i / ICd;
    int o  = t2 % O;
    int r  = t2 / O;
    float v = (ic < ICs) ? fw[((size_t)o * ICs + ic) * 9 + r] : 0.0f;
    w2[i] = (_Float16)v;
}

// ---------------------------------------------------------------------------
// implicit-GEMM conv3x3 via v_wmma_f32_16x16x32_f16, NHWC activations
//   M = 16 pixels, N = 16 out-channels, K = 9 taps x IC (IC % 32 == 0)
//   A fragment: two contiguous 8-half loads per chunk  (NHWC)
//   B fragment: two contiguous 8-half loads per chunk   (weights [9][O][IC])
//   mem NHWC f32 [HW][O]: mem[p][oc] += conv   (pre-decayed by spike kernel)
// ---------------------------------------------------------------------------
__global__ __launch_bounds__(128) void conv3x3_wmma_kernel(
    const _Float16* __restrict__ in,   // [H][W][IC]
    const _Float16* __restrict__ wT,   // [9][O][IC]
    float* __restrict__ mem,           // [HW][O] (+=)
    int IC, int O, int H, int W, int mTiles) {
    const int lane = threadIdx.x & 31;
    const int wave = threadIdx.x >> 5;
    const int tile = blockIdx.x * 4 + wave;
    const int mT = tile % mTiles;
    const int nT = tile / mTiles;
    const bool hi = lane >= 16;
    const int ln  = lane & 15;

    const int p  = mT * 16 + ln;                   // A row (pixel)
    const int py = p / W, px = p - (p / W) * W;
    const int oc = nT * 16 + ln;                   // B column (out channel)

    const int aoff = hi ? 8 : 0;                   // A K-run offset within chunk
    const int boff = hi ? 8 : 0;                   // B K-run offset within chunk
    const v8h vzero = {};

    v8f acc = {};
#pragma unroll
    for (int r = 0; r < 9; ++r) {
        const int dy = r / 3 - 1, dx = r % 3 - 1;  // compile-time (unrolled)
        const int yy = py + dy, xx = px + dx;
        const bool valid = (yy >= 0) & (yy < H) & (xx >= 0) & (xx < W);
        const int yc = min(max(yy, 0), H - 1);
        const int xc = min(max(xx, 0), W - 1);
        const _Float16* abase = in + ((size_t)yc * W + xc) * IC;
        const _Float16* bbase = wT + ((size_t)r * O + oc) * IC;
        for (int kk = 0; kk < IC; kk += 32) {
            __builtin_prefetch(bbase + kk + 32, 0, 0);
            V16H a, b;
            v8h alo = *(const v8h*)(abase + kk + aoff);
            v8h ahi = *(const v8h*)(abase + kk + 16 + aoff);
            a.q[0] = valid ? alo : vzero;
            a.q[1] = valid ? ahi : vzero;
            b.q[0] = *(const v8h*)(bbase + kk + boff);
            b.q[1] = *(const v8h*)(bbase + kk + 16 + boff);
            acc = __builtin_amdgcn_wmma_f32_16x16x32_f16(
                /*neg_a=*/false, a.v, /*neg_b=*/false, b.v,
                /*c_mod=*/(short)0, acc, /*reuse_a=*/false, /*reuse_b=*/false);
        }
    }

    V8F res; res.v = acc;
    // D: VGPR r -> M = r + (hi ? 8 : 0), N = ln ; NHWC store is oc-coalesced
#pragma unroll
    for (int r = 0; r < 8; ++r) {
        int pp = mT * 16 + r + (hi ? 8 : 0);
        mem[(size_t)pp * O + oc] += res.f[r];
    }
}

// ---------------------------------------------------------------------------
// bilinear upsample (half-pixel, edge-clamped) of 5 side accs + 1x1 combine
// ---------------------------------------------------------------------------
__global__ void finalize_kernel(const float* __restrict__ a0,
                                const float* __restrict__ a1,
                                const float* __restrict__ a2,
                                const float* __restrict__ a3,
                                const float* __restrict__ a4,
                                const float* __restrict__ cw,
                                const float* __restrict__ cb,
                                float* __restrict__ out) {
    int p = blockIdx.x * blockDim.x + threadIdx.x;
    if (p >= 128 * 128) return;
    int x = p & 127, y = p >> 7;
    const float* accs[5] = { a0, a1, a2, a3, a4 };
    const int S[5] = { 128, 64, 32, 16, 8 };
    float comb = cb[0];
#pragma unroll
    for (int s = 0; s < 5; ++s) {
        float scale = (float)S[s] * (1.0f / 128.0f);
        float sx = (x + 0.5f) * scale - 0.5f;
        float sy = (y + 0.5f) * scale - 0.5f;
        int x0 = (int)floorf(sx), y0 = (int)floorf(sy);
        float fx = sx - (float)x0, fy = sy - (float)y0;
        int x1 = x0 + 1, y1 = y0 + 1;
        x0 = min(max(x0, 0), S[s] - 1); x1 = min(max(x1, 0), S[s] - 1);
        y0 = min(max(y0, 0), S[s] - 1); y1 = min(max(y1, 0), S[s] - 1);
        const float* A = accs[s];
        float v00 = A[y0 * S[s] + x0], v01 = A[y0 * S[s] + x1];
        float v10 = A[y1 * S[s] + x0], v11 = A[y1 * S[s] + x1];
        float v = v00 * (1.0f - fx) * (1.0f - fy) + v01 * fx * (1.0f - fy)
                + v10 * (1.0f - fx) * fy          + v11 * fx * fy;
        out[s * 16384 + p] = v;
        comb += cw[s] * v;
    }
    out[5 * 16384 + p] = comb;
}

// ---------------------------------------------------------------------------
// host orchestration
// ---------------------------------------------------------------------------
extern "C" void kernel_launch(void* const* d_in, const int* in_sizes, int n_in,
                              void* d_out, int out_size, void* d_ws, size_t ws_size,
                              hipStream_t stream) {
    (void)in_sizes; (void)n_in; (void)out_size; (void)ws_size;

    const float* img  = (const float*)d_in[0];
    const float* rnd  = (const float*)d_in[1];
    const float* fw[13];
    for (int k = 0; k < 13; ++k) fw[k] = (const float*)d_in[2 + k];
    const float* sw[5];
    for (int s = 0; s < 5; ++s) sw[s] = (const float*)d_in[15 + s];
    const float* cw = (const float*)d_in[20];
    const float* cb = (const float*)d_in[21];
    float* out = (float*)d_out;

    // per-layer geometry (IC padded to >=32 for layer 0)
    static const int CH[13]   = {64,64,128,128,256,256,256,512,512,512,512,512,512};
    static const int ICsrc[13]= {3,64,64,128,128,256,256,256,512,512,512,512,512};
    static const int ICd[13]  = {32,64,64,128,128,256,256,256,512,512,512,512,512};
    static const int Hl[13]   = {128,128,64,64,32,32,32,16,16,16,8,8,8};
    static const int ACC_IDX[5] = {1, 3, 6, 9, 12};
    static const int POOL_SRC[4] = {1, 3, 6, 9};

    // workspace layout (256B aligned slabs)
    char* ws = (char*)d_ws;
    size_t cur = 0;
    auto alloc = [&](size_t bytes) -> size_t {
        size_t off = (cur + 255) & ~(size_t)255;
        cur = off + bytes;
        return off;
    };
    float*    memb[13];
    _Float16* outb[13];
    _Float16* wTb[13];
    _Float16* pooled[4];
    float*    accb[5];
    _Float16* actIn;
    int memN[13];
    for (int k = 0; k < 13; ++k) {
        int HW = Hl[k] * Hl[k];
        memN[k] = CH[k] * HW;
        memb[k] = (float*)(ws + alloc((size_t)memN[k] * 4));
        outb[k] = (_Float16*)(ws + alloc((size_t)memN[k] * 2));
        wTb[k]  = (_Float16*)(ws + alloc((size_t)9 * CH[k] * ICd[k] * 2));
    }
    for (int s = 0; s < 4; ++s) {
        int k = POOL_SRC[s];
        int n = CH[k] * (Hl[k] / 2) * (Hl[k] / 2);
        pooled[s] = (_Float16*)(ws + alloc((size_t)n * 2));
    }
    for (int s = 0; s < 5; ++s) {
        int HW = Hl[ACC_IDX[s]] * Hl[ACC_IDX[s]];
        accb[s] = (float*)(ws + alloc((size_t)HW * 4));
    }
    actIn = (_Float16*)(ws + alloc((size_t)16384 * 32 * 2));

    const int B256 = 256;
    auto grid = [](int n) { return (n + 255) / 256; };

    // repack weights f32 -> f16 [9][O][ICd] once per launch
    for (int k = 0; k < 13; ++k) {
        int n = 9 * CH[k] * ICd[k];
        wtrans_kernel<<<grid(n), B256, 0, stream>>>(fw[k], wTb[k], CH[k], ICsrc[k], ICd[k]);
    }
    // zero membranes and side accumulators
    for (int k = 0; k < 13; ++k)
        zero_f32_kernel<<<grid(memN[k]), B256, 0, stream>>>(memb[k], memN[k]);
    for (int s = 0; s < 5; ++s) {
        int HW = Hl[ACC_IDX[s]] * Hl[ACC_IDX[s]];
        zero_f32_kernel<<<grid(HW), B256, 0, stream>>>(accb[s], HW);
    }

    for (int t = 0; t < TSTEPS; ++t) {
        // Poisson spike input, NHWC padded to 32 channels
        poisson_kernel<<<grid(16384 * 32), B256, 0, stream>>>(
            img, rnd + (size_t)t * 3 * 16384, actIn);

        // spikes + membrane pre-decay for every layer (independent)
        for (int k = 0; k < 13; ++k)
            spike_update_kernel<<<grid(memN[k]), B256, 0, stream>>>(memb[k], outb[k], memN[k]);

        // maxpools feeding next stages
        for (int s = 0; s < 4; ++s) {
            int k = POOL_SRC[s];
            int n = CH[k] * (Hl[k] / 2) * (Hl[k] / 2);
            maxpool2_kernel<<<grid(n), B256, 0, stream>>>(outb[k], pooled[s], CH[k], Hl[k], Hl[k]);
        }

        // convs: mem[k] += conv3x3(input_k, w_k)  (independent across layers)
        for (int k = 0; k < 13; ++k) {
            const _Float16* inp;
            if (k == 0)       inp = actIn;
            else if (k == 2)  inp = pooled[0];
            else if (k == 4)  inp = pooled[1];
            else if (k == 7)  inp = pooled[2];
            else if (k == 10) inp = pooled[3];
            else              inp = outb[k - 1];
            int HW = Hl[k] * Hl[k];
            int mTiles = HW / 16;
            int tiles = mTiles * (CH[k] / 16);     // always a multiple of 4
            conv3x3_wmma_kernel<<<tiles / 4, 128, 0, stream>>>(
                inp, wTb[k], memb[k], ICd[k], CH[k], Hl[k], Hl[k], mTiles);
        }

        // side-output accumulation (1x1 conv to 1 channel)
        for (int s = 0; s < 5; ++s) {
            int k = ACC_IDX[s];
            int HW = Hl[k] * Hl[k];
            side_acc_kernel<<<grid(HW), B256, 0, stream>>>(outb[k], sw[s], accb[s], CH[k], HW);
        }
    }

    // upsample + combine -> d_out [6,1,1,128,128]
    finalize_kernel<<<grid(16384), B256, 0, stream>>>(
        accb[0], accb[1], accb[2], accb[3], accb[4], cw, cb, out);
}
